// HeteroProjectionMoEGNN_85495618994898
// MI455X (gfx1250) — matrix operand order repaired and checked by reading, not compiled, not run on hardware
//
#include <hip/hip_runtime.h>
#include <hip/hip_fp16.h>

#define NN   50000
#define EE   800000
#define HIDD 64
#define HEADS 4
#define NEXPT 4

typedef __attribute__((ext_vector_type(16))) _Float16 v16h;
typedef __attribute__((ext_vector_type(8)))  float    v8f;

#define EPI_BIAS 1
#define EPI_RELU 2
#define EPI_SIG  4
#define EPI_ACC  8

#define CDIV(a,b) (((a)+(b)-1)/(b))

// ---------------------------------------------------------------------------
// WMMA GEMM, templated on (K, NC) so the K loop fully unrolls.
// C[M,NC] = epi(A[M,K] @ B[K,NC] (+bias)); f16 inputs, f32 accumulate.
// B (the weight matrix, <=128x64) is staged once per block into LDS,
// transposed [n][k] with a 16-byte row skew, so each lane's B fragment is one
// contiguous 32-byte ds_load. One 16x16 tile per wave, 8 waves/block.
// ---------------------------------------------------------------------------
template<int K, int NC>
__global__ void k_gemm_t(const float* __restrict__ A, int lda,
                         const float* __restrict__ B,          // ldb == NC
                         float* __restrict__ C,                // ldc == NC
                         const float* __restrict__ bias,
                         const float* __restrict__ rowScale, int rsStride,
                         int M, int flags)
{
    constexpr int BSTR = K + 8;                 // half-elements; +16B skew
    __shared__ _Float16 Bs[NC * BSTR];

    // cooperative transpose + f32->f16 convert of the weight matrix
    for (int idx = threadIdx.x; idx < NC * K; idx += blockDim.x) {
        const int n = idx / K, k = idx - n * K;
        Bs[n * BSTR + k] = (_Float16)B[k * NC + n];
    }
    __syncthreads();

    const int lane   = threadIdx.x & 31;
    const int wave   = threadIdx.x >> 5;
    constexpr int colT = NC >> 4;
    const int tiles  = (M >> 4) * colT;
    const int tile   = blockIdx.x * 8 + wave;
    if (tile >= tiles) return;                  // wave-uniform: EXEC all-1 for WMMA
    const int rowTile = tile / colT;
    const int colTile = tile - rowTile * colT;
    const int m0 = rowTile << 4;
    const int n0 = colTile << 4;
    const int half   = lane >> 4;               // K-half select
    const int lane16 = lane & 15;

    v8f acc = {0.f,0.f,0.f,0.f,0.f,0.f,0.f,0.f};
    const float*    Arow = A  + (size_t)(m0 + lane16) * lda;
    const _Float16* Brow = Bs + (n0 + lane16) * BSTR;

#pragma unroll
    for (int kb = 0; kb < K; kb += 32) {
        v16h av, bv;
        // A 16x32 f16 layout: av[8g+t] = Arow[kb + 16*half + 8g + t]
#pragma unroll
        for (int g = 0; g < 2; ++g) {
            const float4* ap = (const float4*)(Arow + kb + 16*half + 8*g);
            const float4 c0 = ap[0], c1 = ap[1];
            av[8*g+0] = (_Float16)c0.x; av[8*g+1] = (_Float16)c0.y;
            av[8*g+2] = (_Float16)c0.z; av[8*g+3] = (_Float16)c0.w;
            av[8*g+4] = (_Float16)c1.x; av[8*g+5] = (_Float16)c1.y;
            av[8*g+6] = (_Float16)c1.z; av[8*g+7] = (_Float16)c1.w;
        }
        // B 32x16 f16 layout: lane n holds K = 16*half + 0..15 -> contiguous in Bs
        bv = *(const v16h*)(Brow + kb + 16*half);
        acc = __builtin_amdgcn_wmma_f32_16x16x32_f16(false, av, false, bv,
                                                     (short)0, acc, false, false);
    }

    const int   cn   = n0 + lane16;
    const float bval = (flags & EPI_BIAS) ? bias[cn] : 0.f;
#pragma unroll
    for (int r = 0; r < 8; ++r) {
        const int cm = m0 + 8*half + r;         // C layout: VGPR r -> M = r + 8*(lane>=16)
        float v = acc[r] + bval;
        if (flags & EPI_RELU) v = fmaxf(v, 0.f);
        if (flags & EPI_SIG)  v = __builtin_amdgcn_rcpf(1.f + __expf(-v));
        if (rowScale)         v *= rowScale[(size_t)cm * rsStride];
        float* cp = C + (size_t)cm * NC + cn;
        if (flags & EPI_ACC) *cp += v; else *cp = v;
    }
}

// ---------------------------------------------------------------------------
// Elementwise / edge kernels
// ---------------------------------------------------------------------------
__global__ void k_fillf(float* p, float v, int n) {
    int i = blockIdx.x * blockDim.x + threadIdx.x; if (i < n) p[i] = v;
}
__global__ void k_fillu(unsigned* p, unsigned v, int n) {
    int i = blockIdx.x * blockDim.x + threadIdx.x; if (i < n) p[i] = v;
}

__global__ void k_select(const float* __restrict__ hp, const float* __restrict__ hd,
                         const int* __restrict__ types,
                         float* hg, float* hi2, int n) {
    int i = blockIdx.x * blockDim.x + threadIdx.x;
    if (i < n) {
        int node = i >> 6;
        float v = (types[node] == 0) ? hp[i] : hd[i];
        hg[i] = v; hi2[i] = v;
    }
}

__global__ void k_deg(const int* __restrict__ dst, float* deg, int nE) {
    int e = blockIdx.x * blockDim.x + threadIdx.x;
    if (e < nE) atomicAdd(&deg[dst[e]], 1.0f);
}

__global__ void k_attn(const float* __restrict__ hW, const float* __restrict__ al,
                       const float* __restrict__ ar, float* el, float* er, int n) {
    int idx = blockIdx.x * blockDim.x + threadIdx.x;
    if (idx < n) {
        int node = idx >> 2, hd = idx & 3;
        const float* p = hW + (size_t)node * HIDD + hd * 16;
        float sl = 0.f, sr = 0.f;
#pragma unroll
        for (int d = 0; d < 16; ++d) { sl += p[d] * al[hd*16+d]; sr += p[d] * ar[hd*16+d]; }
        el[idx] = sl; er[idx] = sr;
    }
}

__device__ __forceinline__ unsigned f2ord(float f) {
    unsigned u = __float_as_uint(f);
    return (u & 0x80000000u) ? ~u : (u | 0x80000000u);
}
__device__ __forceinline__ float ord2f(unsigned u) {
    return (u & 0x80000000u) ? __uint_as_float(u & 0x7fffffffu) : __uint_as_float(~u);
}

__global__ void k_edge_max(const int* __restrict__ src, const int* __restrict__ dst,
                           const float* __restrict__ el, const float* __restrict__ er,
                           unsigned* mE, int nEH) {
    int idx = blockIdx.x * blockDim.x + threadIdx.x;
    if (idx < nEH) {
        int e = idx >> 2, hd = idx & 3;
        float lg = el[src[e]*4+hd] + er[dst[e]*4+hd];
        lg = lg > 0.f ? lg : 0.2f * lg;
        atomicMax(&mE[dst[e]*4+hd], f2ord(lg));
    }
}

__global__ void k_edge_exp(const int* __restrict__ src, const int* __restrict__ dst,
                           const float* __restrict__ el, const float* __restrict__ er,
                           const unsigned* __restrict__ mE, float* ea, float* ss, int nEH) {
    int idx = blockIdx.x * blockDim.x + threadIdx.x;
    if (idx < nEH) {
        int e = idx >> 2, hd = idx & 3;
        int d = dst[e];
        float lg = el[src[e]*4+hd] + er[d*4+hd];
        lg = lg > 0.f ? lg : 0.2f * lg;
        float ev = __expf(lg - ord2f(mE[d*4+hd]));
        ea[idx] = ev;
        atomicAdd(&ss[d*4+hd], ev);
    }
}

__global__ void k_edge_agg(const int* __restrict__ src, const int* __restrict__ dst,
                           const float* __restrict__ hW, const float* __restrict__ ea,
                           const float* __restrict__ ss, float* agg, int nEH) {
    int idx = blockIdx.x * blockDim.x + threadIdx.x;
    if (idx < nEH) {
        int e = idx >> 2, hd = idx & 3;
        int d = dst[e], s = src[e];
        float a = ea[idx] / (ss[d*4+hd] + 1e-9f);
        const float* hp = hW + (size_t)s * HIDD + hd * 16;
        __builtin_prefetch(hp, 0, 1);
        float* op = agg + (size_t)d * HIDD + hd * 16;
#pragma unroll
        for (int k = 0; k < 16; ++k) atomicAdd(&op[k], hp[k] * a);
    }
}

__global__ void k_gat_epi(float* hgat, const float* __restrict__ agg, int n, int doRelu) {
    int i = blockIdx.x * blockDim.x + threadIdx.x;
    if (i < n) {
        float v = hgat[i] + agg[i];
        hgat[i] = doRelu ? fmaxf(v, 0.f) : v;
    }
}

__global__ void k_gin_agg(const int* __restrict__ src, const int* __restrict__ dst,
                          const float* __restrict__ h, float* agg, int nEC) {
    int idx = blockIdx.x * blockDim.x + threadIdx.x;
    if (idx < nEC) {
        int e = idx >> 6, c = idx & 63;
        atomicAdd(&agg[(size_t)dst[e]*HIDD + c], h[(size_t)src[e]*HIDD + c]);
    }
}

__global__ void k_gin_x(const float* __restrict__ h, const float* __restrict__ agg,
                        const float* __restrict__ deg, const float* __restrict__ eps,
                        int l, float* x, int n) {
    int i = blockIdx.x * blockDim.x + threadIdx.x;
    if (i < n) {
        int node = i >> 6;
        float iv = 1.f / fmaxf(deg[node], 1.f);
        x[i] = (1.f + eps[l]) * h[i] + agg[i] * iv;
    }
}

// gating: softmax over 4 logits, top-2 renormalized, stats for aux loss
__global__ void k_gate(const float* __restrict__ z, const float* __restrict__ gW,
                       const float* __restrict__ gb, float* wgt,
                       float* fsum, float* Psum, int n) {
    int node = blockIdx.x * blockDim.x + threadIdx.x;
    if (node >= n) return;
    float lg[4] = {gb[0], gb[1], gb[2], gb[3]};
    const float* zp = z + (size_t)node * HIDD;
    for (int d = 0; d < HIDD; ++d) {
        float zv = zp[d];
#pragma unroll
        for (int j = 0; j < 4; ++j) lg[j] += zv * gW[d*4+j];
    }
    float m = lg[0];
#pragma unroll
    for (int j = 1; j < 4; ++j) m = fmaxf(m, lg[j]);
    float p[4], s = 0.f;
#pragma unroll
    for (int j = 0; j < 4; ++j) { p[j] = __expf(lg[j] - m); s += p[j]; }
    float inv = 1.f / s;
#pragma unroll
    for (int j = 0; j < 4; ++j) p[j] *= inv;
    int i0 = 0;
#pragma unroll
    for (int j = 1; j < 4; ++j) if (p[j] > p[i0]) i0 = j;
    int i1 = -1;
#pragma unroll
    for (int j = 0; j < 4; ++j) if (j != i0 && (i1 < 0 || p[j] > p[i1])) i1 = j;
    float tinv = 1.f / (p[i0] + p[i1]);
    float w[4] = {0.f, 0.f, 0.f, 0.f};
    w[i0] = p[i0] * tinv; w[i1] = p[i1] * tinv;
#pragma unroll
    for (int j = 0; j < 4; ++j) wgt[(size_t)node*4 + j] = w[j];
    atomicAdd(&fsum[i0], 1.f);
    atomicAdd(&fsum[i1], 1.f);
#pragma unroll
    for (int j = 0; j < 4; ++j) atomicAdd(&Psum[j], p[j]);
}

__global__ void k_concat(const float* __restrict__ zg, const float* __restrict__ zi,
                         float* u, int n) {
    int i = blockIdx.x * blockDim.x + threadIdx.x;
    if (i < n) {
        int node = i >> 7, c = i & 127;
        u[i] = (c < 64) ? zg[(size_t)node*64 + c] : zi[(size_t)node*64 + (c - 64)];
    }
}

__global__ void k_final(const float* __restrict__ gate, const float* __restrict__ zg,
                        const float* __restrict__ zi, float* out, int n) {
    int i = blockIdx.x * blockDim.x + threadIdx.x;
    if (i < n) {
        float g = gate[i];
        float z = zg[i] * zi[i];
        out[i] = g * z + (1.f - g) * g;   // u_prime == gate (deterministic double-call)
    }
}

__global__ void k_aux(const float* __restrict__ stats, float* out) {
    if (blockIdx.x == 0 && threadIdx.x == 0) {
        float a = 0.f;
        for (int b = 0; b < 2; ++b) {
            const float* f = stats + b*8;
            const float* P = stats + b*8 + 4;
            float s = 0.f;
            for (int j = 0; j < 4; ++j)
                s += (f[j] / (float)NN) * (P[j] / (float)NN);
            a += (float)NEXPT * s;
        }
        out[0] = a;
    }
}

// ---------------------------------------------------------------------------
// Host-side launch helpers
// ---------------------------------------------------------------------------
static void gemm(hipStream_t s, int K, int NC,
                 const float* A, int lda, const float* B,
                 float* C, const float* bias, const float* rs, int rss,
                 int M, int flags) {
    const int tiles  = (M / 16) * (NC / 16);
    const int blocks = CDIV(tiles, 8);
    if      (K == 128 && NC == 64)
        k_gemm_t<128,64><<<blocks,256,0,s>>>(A,lda,B,C,bias,rs,rss,M,flags);
    else if (K == 64  && NC == 64)
        k_gemm_t<64,64><<<blocks,256,0,s>>>(A,lda,B,C,bias,rs,rss,M,flags);
    else if (K == 32  && NC == 64)
        k_gemm_t<32,64><<<blocks,256,0,s>>>(A,lda,B,C,bias,rs,rss,M,flags);
    else if (K == 64  && NC == 32)
        k_gemm_t<64,32><<<blocks,256,0,s>>>(A,lda,B,C,bias,rs,rss,M,flags);
}

static void expert_block(hipStream_t s, const float* z,
                         const float* gW, const float* gb,
                         const float* eW1, const float* eb1,
                         const float* eW2, const float* eb2,
                         float* zout, float* wgt, float* t32,
                         float* fsum, float* Psum) {
    k_gate<<<CDIV(NN, 256), 256, 0, s>>>(z, gW, gb, wgt, fsum, Psum, NN);
    for (int e = 0; e < NEXPT; ++e) {
        // e1 = relu(z @ eW1[e] + eb1[e])   [N,32]
        gemm(s, 64, 32, z, 64, eW1 + (size_t)e*64*32, t32, eb1 + e*32,
             nullptr, 0, NN, EPI_BIAS | EPI_RELU);
        // zout (+)= w[:,e] * (e1 @ eW2[e] + eb2[e])   [N,64]
        gemm(s, 32, 64, t32, 32, eW2 + (size_t)e*32*64, zout, eb2 + e*64,
             wgt + e, 4, NN, EPI_BIAS | (e ? EPI_ACC : 0));
    }
}

extern "C" void kernel_launch(void* const* d_in, const int* in_sizes, int n_in,
                              void* d_out, int out_size, void* d_ws, size_t ws_size,
                              hipStream_t stream) {
    const float* features  = (const float*)d_in[0];
    const int*   node_types= (const int*)  d_in[1];
    const int*   src       = (const int*)  d_in[2];
    const int*   dst       = (const int*)  d_in[3];
    const float* W_person  = (const float*)d_in[4];
    const float* b_person  = (const float*)d_in[5];
    const float* W_disease = (const float*)d_in[6];
    const float* b_disease = (const float*)d_in[7];
    const float* gat_W     = (const float*)d_in[8];
    const float* gat_al    = (const float*)d_in[9];
    const float* gat_ar    = (const float*)d_in[10];
    const float* gin_eps   = (const float*)d_in[11];
    const float* gin_W1    = (const float*)d_in[12];
    const float* gin_b1    = (const float*)d_in[13];
    const float* gin_W2    = (const float*)d_in[14];
    const float* gin_b2    = (const float*)d_in[15];
    const float* gat_gW    = (const float*)d_in[16];
    const float* gat_gb    = (const float*)d_in[17];
    const float* gat_eW1   = (const float*)d_in[18];
    const float* gat_eb1   = (const float*)d_in[19];
    const float* gat_eW2   = (const float*)d_in[20];
    const float* gat_eb2   = (const float*)d_in[21];
    const float* gin_gW    = (const float*)d_in[22];
    const float* gin_gb    = (const float*)d_in[23];
    const float* gin_eW1   = (const float*)d_in[24];
    const float* gin_eb1   = (const float*)d_in[25];
    const float* gin_eW2   = (const float*)d_in[26];
    const float* gin_eb2   = (const float*)d_in[27];
    const float* fg_W1     = (const float*)d_in[28];
    const float* fg_b1     = (const float*)d_in[29];
    const float* fg_W2     = (const float*)d_in[30];
    const float* fg_b2     = (const float*)d_in[31];
    (void)in_sizes; (void)n_in; (void)out_size; (void)ws_size;

    // ---- workspace carve-up (floats) ----
    float* w = (float*)d_ws;
    size_t off = 0;
    auto alloc = [&](size_t n) { float* p = w + off; off += n; return p; };
    float* hgat = alloc((size_t)NN * 64);
    float* hgin = alloc((size_t)NN * 64);
    float* hW   = alloc((size_t)NN * 64);
    float* xbuf = alloc((size_t)NN * 64);
    float* aggb = alloc((size_t)NN * 64);
    float* tbuf = alloc((size_t)NN * 64);
    float* zg   = alloc((size_t)NN * 64);
    float* zi   = alloc((size_t)NN * 64);
    float* gate = alloc((size_t)NN * 64);
    float* ubuf = alloc((size_t)NN * 128);
    float* t32  = alloc((size_t)NN * 32);
    float* ea   = alloc((size_t)EE * 4);
    float* el   = alloc((size_t)NN * 4);
    float* er   = alloc((size_t)NN * 4);
    float* ss   = alloc((size_t)NN * 4);
    unsigned* mE = (unsigned*)alloc((size_t)NN * 4);
    float* deg  = alloc((size_t)NN);
    float* wgt  = alloc((size_t)NN * 4);
    float* stats= alloc(16);

    const int NH   = NN * 64;
    const int EH   = EE * HEADS;
    const int NEHD = NN * HEADS;
    const unsigned NEG_INF_ORD = 0x007FFFFFu;   // f2ord(-inf)

    // ---- 1. per-type projection: hp=tbuf, hd=aggb, then select into hgat/hgin
    gemm(stream, 128, 64, features, 128, W_person,  tbuf, b_person,  nullptr, 0, NN, EPI_BIAS);
    gemm(stream,  64, 64, features, 128, W_disease, aggb, b_disease, nullptr, 0, NN, EPI_BIAS);
    k_select<<<CDIV(NH,256),256,0,stream>>>(tbuf, aggb, node_types, hgat, hgin, NH);

    // ---- degree (GIN mean aggregator)
    k_fillf<<<CDIV(NN,256),256,0,stream>>>(deg, 0.f, NN);
    k_deg  <<<CDIV(EE,256),256,0,stream>>>(dst, deg, EE);

    // ---- 2. GAT (2 layers)
    for (int l = 0; l < 2; ++l) {
        gemm(stream, 64, 64, hgat, 64, gat_W + (size_t)l*64*64, hW,
             nullptr, nullptr, 0, NN, 0);
        k_attn<<<CDIV(NEHD,256),256,0,stream>>>(hW, gat_al + l*HEADS*16,
                                                gat_ar + l*HEADS*16, el, er, NEHD);
        k_fillu<<<CDIV(NEHD,256),256,0,stream>>>(mE, NEG_INF_ORD, NEHD);
        k_fillf<<<CDIV(NEHD,256),256,0,stream>>>(ss, 0.f, NEHD);
        k_fillf<<<CDIV(NH,256),256,0,stream>>>(aggb, 0.f, NH);
        k_edge_max<<<CDIV(EH,256),256,0,stream>>>(src, dst, el, er, mE, EH);
        k_edge_exp<<<CDIV(EH,256),256,0,stream>>>(src, dst, el, er, mE, ea, ss, EH);
        k_edge_agg<<<CDIV(EH,256),256,0,stream>>>(src, dst, hW, ea, ss, aggb, EH);
        k_gat_epi<<<CDIV(NH,256),256,0,stream>>>(hgat, aggb, NH, (l == 0) ? 1 : 0);
    }

    // ---- 3. GIN (2 layers)
    for (int l = 0; l < 2; ++l) {
        k_fillf<<<CDIV(NH,256),256,0,stream>>>(aggb, 0.f, NH);
        k_gin_agg<<<CDIV(EE*64,256),256,0,stream>>>(src, dst, hgin, aggb, EE*64);
        k_gin_x<<<CDIV(NH,256),256,0,stream>>>(hgin, aggb, deg, gin_eps, l, xbuf, NH);
        gemm(stream, 64, 64, xbuf, 64, gin_W1 + (size_t)l*64*64, tbuf,
             gin_b1 + l*64, nullptr, 0, NN, EPI_BIAS | EPI_RELU);
        gemm(stream, 64, 64, tbuf, 64, gin_W2 + (size_t)l*64*64, hgin,
             gin_b2 + l*64, nullptr, 0, NN,
             EPI_BIAS | ((l == 0) ? EPI_RELU : 0));
    }

    // ---- 4. MoE experts (zg from GAT branch, zi from GIN branch)
    k_fillf<<<1,64,0,stream>>>(stats, 0.f, 16);
    expert_block(stream, hgat, gat_gW, gat_gb, gat_eW1, gat_eb1, gat_eW2, gat_eb2,
                 zg, wgt, t32, stats + 0, stats + 4);
    expert_block(stream, hgin, gin_gW, gin_gb, gin_eW1, gin_eb1, gin_eW2, gin_eb2,
                 zi, wgt, t32, stats + 8, stats + 12);

    // ---- 5. fusion gate
    k_concat<<<CDIV(NN*128,256),256,0,stream>>>(zg, zi, ubuf, NN*128);
    gemm(stream, 128, 64, ubuf, 128, fg_W1, tbuf, fg_b1, nullptr, 0, NN,
         EPI_BIAS | EPI_RELU);
    gemm(stream,  64, 64, tbuf,  64, fg_W2, gate, fg_b2, nullptr, 0, NN,
         EPI_BIAS | EPI_SIG);

    // ---- 6. output
    k_final<<<CDIV(NH,256),256,0,stream>>>(gate, zg, zi, (float*)d_out, NH);
    k_aux<<<1,1,0,stream>>>(stats, (float*)d_out + (size_t)NN*64);
}
